// MultiHeadAttention_48739288875398
// MI455X (gfx1250) — compile-verified, matrix operand order checked by reading, not run on hardware
//
#include <hip/hip_runtime.h>

typedef __attribute__((ext_vector_type(16))) _Float16 v16h;
typedef __attribute__((ext_vector_type(8)))  _Float16 v8h;
typedef __attribute__((ext_vector_type(4)))  _Float16 v4h;
typedef __attribute__((ext_vector_type(8)))  float    v8f;
typedef __attribute__((ext_vector_type(4)))  float    v4f;

#define DEV __device__ __forceinline__

constexpr int B_ = 8, S_ = 2048, D_ = 512, H_ = 8, DH_ = 64;
constexpr int M_ = B_ * S_;                 // 16384 rows

// ---------------------------------------------------------------------------
// Fragment loader for 16-bit WMMA operands (A: lane = M row, B: lane = N col).
// ISA layout (CDNA5 ISA §7.12.2): lanes 0-15 hold K = {k0..k0+7,
// k0+16..k0+23}; lanes 16-31 hold K = {k0+8..k0+15, k0+24..k0+31}. Both
// 8-half chunks are contiguous 16-byte loads -> two global_load_b128 / lane.
// ---------------------------------------------------------------------------
DEV v16h load_frag(const _Float16* __restrict__ base, int ld, int row0, int k0, int lane) {
  const int r  = row0 + (lane & 15);
  const int hb = (lane >> 4) << 3;                 // 0 or 8
  const _Float16* p = base + (size_t)r * ld + (k0 + hb);
  v8h lo = *(const v8h*)(p);
  v8h hi = *(const v8h*)(p + 16);
  v16h f;
#pragma unroll
  for (int i = 0; i < 8; ++i) { f[i] = lo[i]; f[i + 8] = hi[i]; }
  return f;
}

DEV v8f wmma_f16(v16h a, v16h b, v8f c) {
  return __builtin_amdgcn_wmma_f32_16x16x32_f16(false, a, false, b, (short)0, c,
                                                false, false);
}

// ---------------------------------------------------------------------------
// fp32 -> fp16 conversion (vectorized x4)
// ---------------------------------------------------------------------------
__global__ void __launch_bounds__(256)
cvt_f32_f16(const float* __restrict__ in, _Float16* __restrict__ out, int n4) {
  int i = blockIdx.x * blockDim.x + threadIdx.x;
  if (i < n4) {
    v4f x = ((const v4f*)in)[i];
    v4h y;
    y[0] = (_Float16)x[0]; y[1] = (_Float16)x[1];
    y[2] = (_Float16)x[2]; y[3] = (_Float16)x[3];
    ((v4h*)out)[i] = y;
  }
}

// ---------------------------------------------------------------------------
// C = (A @ W^T + bias) * scale.  A:[M,K] f16 row-major, W:[N,K] f16 row-major.
// Block = 256 thr (8 waves), tile 128x128, wave tile 32x64 (2x4 accumulators).
// mode 0: f16 [M,N]   mode 1: f16 [B,H,S,DH]   mode 2: f16 [B,H,DH,S]
// mode 3: f32 [M,N] (final output)
// ---------------------------------------------------------------------------
__global__ void __launch_bounds__(256)
gemm_wmma(const _Float16* __restrict__ A, const _Float16* __restrict__ W,
          const float* __restrict__ bias, void* __restrict__ out,
          int M, int N, int K, float scale, int mode) {
  const int lane = threadIdx.x & 31;
  const int wid  = threadIdx.x >> 5;
  const int tm = blockIdx.x * 128 + (wid & 3) * 32;
  const int tn = blockIdx.y * 128 + (wid >> 2) * 64;

  const _Float16* Arow = A + (size_t)tm * K;
  const _Float16* Wrow = W + (size_t)tn * K;

  v8f acc[2][4] = {};
#pragma unroll 2
  for (int k0 = 0; k0 < K; k0 += 32) {
    v16h a0 = load_frag(Arow, K, 0,  k0, lane);
    v16h a1 = load_frag(Arow, K, 16, k0, lane);
    v16h b0 = load_frag(Wrow, K, 0,  k0, lane);
    v16h b1 = load_frag(Wrow, K, 16, k0, lane);
    v16h b2 = load_frag(Wrow, K, 32, k0, lane);
    v16h b3 = load_frag(Wrow, K, 48, k0, lane);
    acc[0][0] = wmma_f16(a0, b0, acc[0][0]);
    acc[0][1] = wmma_f16(a0, b1, acc[0][1]);
    acc[0][2] = wmma_f16(a0, b2, acc[0][2]);
    acc[0][3] = wmma_f16(a0, b3, acc[0][3]);
    acc[1][0] = wmma_f16(a1, b0, acc[1][0]);
    acc[1][1] = wmma_f16(a1, b1, acc[1][1]);
    acc[1][2] = wmma_f16(a1, b2, acc[1][2]);
    acc[1][3] = wmma_f16(a1, b3, acc[1][3]);
  }

  // C layout: VGPR r, lanes0-15 -> M=r,N=lane; lanes16-31 -> M=r+8,N=lane-16
  const int hi = lane >> 4, ln = lane & 15;
#pragma unroll
  for (int mi = 0; mi < 2; ++mi)
#pragma unroll
    for (int j = 0; j < 4; ++j) {
      const int n  = tn + 16 * j + ln;
      const float bv = bias[n];
#pragma unroll
      for (int r = 0; r < 8; ++r) {
        const int m = tm + 16 * mi + (hi << 3) + r;
        const float v = (acc[mi][j][r] + bv) * scale;
        if (mode == 0) {
          ((_Float16*)out)[(size_t)m * N + n] = (_Float16)v;
        } else if (mode == 1) {                    // [B,H,S,DH]
          const int bb = m >> 11, s = m & (S_ - 1), hh = n >> 6, dh = n & 63;
          ((_Float16*)out)[(((size_t)bb * H_ + hh) * S_ + s) * DH_ + dh] = (_Float16)v;
        } else if (mode == 2) {                    // [B,H,DH,S] (V transposed)
          const int bb = m >> 11, s = m & (S_ - 1), hh = n >> 6, dh = n & 63;
          ((_Float16*)out)[(((size_t)bb * H_ + hh) * DH_ + dh) * S_ + s] = (_Float16)v;
        } else {                                   // f32 final
          ((float*)out)[(size_t)m * N + n] = v;
        }
      }
    }
}

// ---------------------------------------------------------------------------
// Flash attention: block = 128 thr (4 waves), each wave = 16 query rows.
// grid.x = B*H, grid.y = S/64. Online softmax fp32; key mask -1e9 (== ref).
// ---------------------------------------------------------------------------
__global__ void __launch_bounds__(128)
attn_wmma(const _Float16* __restrict__ Qh, const _Float16* __restrict__ Kh,
          const _Float16* __restrict__ Vt, const int* __restrict__ lens,
          _Float16* __restrict__ ctx) {
  __shared__ _Float16 Plds[4][16 * 64];            // 2 KB per wave
  const int lane = threadIdx.x & 31;
  const int wid  = threadIdx.x >> 5;
  const int bh = blockIdx.x;
  const int b  = bh >> 3;
  const int h  = bh & (H_ - 1);
  const int q0 = blockIdx.y * 64 + wid * 16;

  const _Float16* Qbh = Qh + (size_t)bh * S_ * DH_;
  const _Float16* Kbh = Kh + (size_t)bh * S_ * DH_;
  const _Float16* Vbh = Vt + (size_t)bh * DH_ * S_;
  const int len  = lens[b];
  const int kend = (len + 63) & ~63;               // skip fully-masked chunks

  v16h qf0 = load_frag(Qbh + (size_t)q0 * DH_, DH_, 0, 0,  lane);
  v16h qf1 = load_frag(Qbh + (size_t)q0 * DH_, DH_, 0, 32, lane);

  v8f cacc[4] = {};
  float rm[8], rs[8];
#pragma unroll
  for (int r = 0; r < 8; ++r) { rm[r] = -3.0e38f; rs[r] = 0.f; }

  const int hi = lane >> 4, ln = lane & 15;
  constexpr float L2E = 1.4426950408889634f;
  _Float16* pw = &Plds[wid][0];

  for (int kv = 0; kv < kend; kv += 64) {
    // S = Q K^T : 4 x (16x16) tiles, K fragments are K rows (row-contiguous)
    v8f sf[4];
#pragma unroll
    for (int j = 0; j < 4; ++j) {
      v16h kf0 = load_frag(Kbh + (size_t)(kv + 16 * j) * DH_, DH_, 0, 0,  lane);
      v16h kf1 = load_frag(Kbh + (size_t)(kv + 16 * j) * DH_, DH_, 0, 32, lane);
      v8f c = {};
      c = wmma_f16(qf0, kf0, c);
      c = wmma_f16(qf1, kf1, c);
      sf[j] = c;
    }
    // key padding mask (column-only -> uniform over the 8 row-VGPRs)
#pragma unroll
    for (int j = 0; j < 4; ++j)
      if (kv + 16 * j + ln >= len) {
#pragma unroll
        for (int r = 0; r < 8; ++r) sf[j][r] = -1.0e9f;
      }
    // online softmax: row max/sum live across the 16-lane N group
#pragma unroll
    for (int r = 0; r < 8; ++r) {
      float v = fmaxf(fmaxf(sf[0][r], sf[1][r]), fmaxf(sf[2][r], sf[3][r]));
      v = fmaxf(v, __shfl_xor(v, 1));
      v = fmaxf(v, __shfl_xor(v, 2));
      v = fmaxf(v, __shfl_xor(v, 4));
      v = fmaxf(v, __shfl_xor(v, 8));
      const float nm   = fmaxf(rm[r], v);
      const float corr = __builtin_amdgcn_exp2f((rm[r] - nm) * L2E);
      rs[r] *= corr;
#pragma unroll
      for (int j = 0; j < 4; ++j) cacc[j][r] *= corr;
      rm[r] = nm;
    }
#pragma unroll
    for (int r = 0; r < 8; ++r) {
      float part = 0.f;
#pragma unroll
      for (int j = 0; j < 4; ++j) {
        const float p = __builtin_amdgcn_exp2f((sf[j][r] - rm[r]) * L2E);
        sf[j][r] = p;
        part += p;
      }
      part += __shfl_xor(part, 1);
      part += __shfl_xor(part, 2);
      part += __shfl_xor(part, 4);
      part += __shfl_xor(part, 8);
      rs[r] += part;
    }
    // transpose P (C-layout -> A-layout) through per-wave LDS
#pragma unroll
    for (int j = 0; j < 4; ++j)
#pragma unroll
      for (int r = 0; r < 8; ++r)
        pw[((hi << 3) + r) * 64 + 16 * j + ln] = (_Float16)sf[j][r];
    v16h pf0 = load_frag(pw, 64, 0, 0,  lane);
    v16h pf1 = load_frag(pw, 64, 0, 32, lane);
    // ctx += P @ V : V fragments from transposed V (rows = dh, k = s)
#pragma unroll
    for (int j = 0; j < 4; ++j) {
      v16h vf0 = load_frag(Vbh, S_, 16 * j, kv,      lane);
      v16h vf1 = load_frag(Vbh, S_, 16 * j, kv + 32, lane);
      cacc[j] = wmma_f16(pf0, vf0, cacc[j]);
      cacc[j] = wmma_f16(pf1, vf1, cacc[j]);
    }
  }
  // normalize and store to ctx [B, S, H*DH] f16
#pragma unroll
  for (int r = 0; r < 8; ++r) {
    const float inv = 1.0f / rs[r];
#pragma unroll
    for (int j = 0; j < 4; ++j) cacc[j][r] *= inv;
  }
  _Float16* cb = ctx + ((size_t)b * S_ + q0) * (H_ * DH_);
#pragma unroll
  for (int j = 0; j < 4; ++j) {
    const int n = h * DH_ + 16 * j + ln;
#pragma unroll
    for (int r = 0; r < 8; ++r)
      cb[(size_t)((hi << 3) + r) * (H_ * DH_) + n] = (_Float16)cacc[j][r];
  }
}

// ---------------------------------------------------------------------------
extern "C" void kernel_launch(void* const* d_in, const int* in_sizes, int n_in,
                              void* d_out, int out_size, void* d_ws, size_t ws_size,
                              hipStream_t stream) {
  const float* xQ = (const float*)d_in[0];
  const float* xK = (const float*)d_in[1];
  const float* xV = (const float*)d_in[2];
  const float* Wq = (const float*)d_in[3];
  const float* bq = (const float*)d_in[4];
  const float* Wk = (const float*)d_in[5];
  const float* bk = (const float*)d_in[6];
  const float* Wv = (const float*)d_in[7];
  const float* bv = (const float*)d_in[8];
  const float* Wo = (const float*)d_in[9];
  const float* bo = (const float*)d_in[10];
  const int* lens = (const int*)d_in[11];

  char* ws = (char*)d_ws;
  const size_t szX = (size_t)M_ * D_ * sizeof(_Float16);   // 16 MiB
  const size_t szW = (size_t)D_ * D_ * sizeof(_Float16);   // 0.5 MiB
  _Float16* xqh = (_Float16*)(ws);
  _Float16* xkh = (_Float16*)(ws + szX);
  _Float16* xvh = (_Float16*)(ws + 2 * szX);
  _Float16* wqh = (_Float16*)(ws + 3 * szX);
  _Float16* wkh = (_Float16*)(ws + 3 * szX + szW);
  _Float16* wvh = (_Float16*)(ws + 3 * szX + 2 * szW);
  _Float16* woh = (_Float16*)(ws + 3 * szX + 3 * szW);
  _Float16* Qh  = (_Float16*)(ws + 3 * szX + 4 * szW);
  _Float16* Kh  = (_Float16*)(ws + 4 * szX + 4 * szW);
  _Float16* Vt  = (_Float16*)(ws + 5 * szX + 4 * szW);
  _Float16* ctxh = xqh;   // overlay: x_Q f16 is dead after the Q projection

  const int nX4 = M_ * D_ / 4;
  const int nW4 = D_ * D_ / 4;
  cvt_f32_f16<<<(nX4 + 255) / 256, 256, 0, stream>>>(xQ, xqh, nX4);
  cvt_f32_f16<<<(nX4 + 255) / 256, 256, 0, stream>>>(xK, xkh, nX4);
  cvt_f32_f16<<<(nX4 + 255) / 256, 256, 0, stream>>>(xV, xvh, nX4);
  cvt_f32_f16<<<(nW4 + 255) / 256, 256, 0, stream>>>(Wq, wqh, nW4);
  cvt_f32_f16<<<(nW4 + 255) / 256, 256, 0, stream>>>(Wk, wkh, nW4);
  cvt_f32_f16<<<(nW4 + 255) / 256, 256, 0, stream>>>(Wv, wvh, nW4);
  cvt_f32_f16<<<(nW4 + 255) / 256, 256, 0, stream>>>(Wo, woh, nW4);

  dim3 gg(M_ / 128, D_ / 128);   // (128, 4)
  gemm_wmma<<<gg, 256, 0, stream>>>(xqh, wqh, bq, Qh, M_, D_, D_, 0.125f, 1);
  gemm_wmma<<<gg, 256, 0, stream>>>(xkh, wkh, bk, Kh, M_, D_, D_, 1.0f,   1);
  gemm_wmma<<<gg, 256, 0, stream>>>(xvh, wvh, bv, Vt, M_, D_, D_, 1.0f,   2);

  attn_wmma<<<dim3(B_ * H_, S_ / 64), 128, 0, stream>>>(Qh, Kh, Vt, lens, ctxh);

  gemm_wmma<<<gg, 256, 0, stream>>>(ctxh, woh, bo, d_out, M_, D_, D_, 1.0f, 3);
}
